// SRN_36893769073101
// MI455X (gfx1250) — matrix operand order repaired and checked by reading, not compiled, not run on hardware
//
#include <hip/hip_runtime.h>

typedef __bf16 bf16;
typedef __attribute__((ext_vector_type(16))) __bf16 v16bf;
typedef __attribute__((ext_vector_type(8)))  float  v8f;

#define B_  1024
#define Q_  32
#define D_  300
#define A_  256
#define DP  320   // D padded to 10 * 32 (K-chunks of 32 for bf16 WMMA)
#define EPSF 2.220446049250313e-16f

// ---- LDS byte offsets (dynamic shared memory partition) ----
#define OFF_QA     0         // bf16 [256][320]: gathered r_emb, later q_attn (163840 B)
#define OFF_QB     163840    // bf16 [32][320] : q_t row-major                (20480 B)
#define OFF_QTB    184320    // bf16 [320][32] : q_t col-major                (20480 B)
#define OFF_LOG    204800    // f32  [256][32] : logits                       (32768 B)
#define OFF_ATT    237568    // bf16 [256][32] : attention weights            (16384 B)
#define OFF_PAN    253952    // bf16 [64][32]  : W2 panel, transposed         (4096 B)
#define OFF_BASE   258048    // f32  [300]     : last_h @ W1 + b_ac           (1280 B)
#define OFF_WB     259328    // bf16 [320]     : w_rel padded                 (640 B)
#define OFF_SC     259968    // f32  [256]     : scores                       (1024 B)
#define OFF_RIDX   260992    // i32  [256]     : gathered indices             (1024 B)
#define OFF_RED    262016    // f32  [8]       : block reduction scratch      (32 B)
#define SMEM_BYTES 262144

// A-fragment (16x32 bf16) from row-major LDS, ISA 7.12.2 layout:
// lane = 16*hi + m ; elems 0..7 -> K = k0+8*hi+0..7 ; elems 8..15 -> K = k0+16+8*hi+0..7
__device__ __forceinline__ v16bf frag_a_lds(const bf16* __restrict__ base,
                                            int row0, int k0, int stride, int lane) {
  const int m = lane & 15, hi = lane >> 4;
  const bf16* p = base + (row0 + m) * stride + k0 + 8 * hi;
  v16bf f;
#pragma unroll
  for (int i = 0; i < 8; ++i) { f[i] = p[i]; f[8 + i] = p[16 + i]; }
  return f;
}

// B-fragment (32x16 bf16) where element (k,n) is stored at baseT[n*strideT + k]:
// lane = 16*hi + n ; elems 0..7 -> K = k0+8*hi+0..7 ; elems 8..15 -> K = k0+16+8*hi+0..7
__device__ __forceinline__ v16bf frag_b_lds(const bf16* __restrict__ baseT,
                                            int n0, int k0, int strideT, int lane) {
  const int n = lane & 15, hi = lane >> 4;
  const bf16* p = baseT + (n0 + n) * strideT + k0 + 8 * hi;
  v16bf f;
#pragma unroll
  for (int i = 0; i < 8; ++i) { f[i] = p[i]; f[8 + i] = p[16 + i]; }
  return f;
}

__device__ __forceinline__ float block_max(float v, float* red, int tid) {
#pragma unroll
  for (int m = 16; m >= 1; m >>= 1) v = fmaxf(v, __shfl_xor(v, m, 32));
  if ((tid & 31) == 0) red[tid >> 5] = v;
  __syncthreads();
  float r = red[0];
#pragma unroll
  for (int i = 1; i < 8; ++i) r = fmaxf(r, red[i]);
  __syncthreads();
  return r;
}

__device__ __forceinline__ float block_sum(float v, float* red, int tid) {
#pragma unroll
  for (int m = 16; m >= 1; m >>= 1) v += __shfl_xor(v, m, 32);
  if ((tid & 31) == 0) red[tid >> 5] = v;
  __syncthreads();
  float r = 0.f;
#pragma unroll
  for (int i = 0; i < 8; ++i) r += red[i];
  __syncthreads();
  return r;
}

__global__ void __launch_bounds__(256)
srn_policy_kernel(const float* __restrict__ q_t,     // [B,Q,D]
                  const float* __restrict__ last_h,  // [B,D]
                  const float* __restrict__ rel,     // [R,D]
                  const float* __restrict__ w_rel,   // [D]
                  const float* __restrict__ b_rel,   // [1]
                  const float* __restrict__ W_ac,    // [2D,D]
                  const float* __restrict__ b_ac,    // [D]
                  const int*   __restrict__ r_space, // [B,A]
                  const float* __restrict__ amask,   // [B,A]
                  float*       __restrict__ out)     // [B,A]
{
  extern __shared__ char smem[];
  bf16*  qa     = (bf16*)(smem + OFF_QA);
  bf16*  qb     = (bf16*)(smem + OFF_QB);
  bf16*  qTb    = (bf16*)(smem + OFF_QTB);
  float* logits = (float*)(smem + OFF_LOG);
  bf16*  attnb  = (bf16*)(smem + OFF_ATT);
  bf16*  panelT = (bf16*)(smem + OFF_PAN);
  float* basev  = (float*)(smem + OFF_BASE);
  bf16*  wb     = (bf16*)(smem + OFF_WB);
  float* s_sc   = (float*)(smem + OFF_SC);
  int*   ridx   = (int*)(smem + OFF_RIDX);
  float* red    = (float*)(smem + OFF_RED);

  const int tid  = threadIdx.x;
  const int lane = tid & 31;
  const int wv   = tid >> 5;      // 8 waves
  const int b    = blockIdx.x;
  const int nl   = lane & 15;
  const int hi   = lane >> 4;

  // ---------------- Phase 0: staging ----------------
  s_sc[tid] = 0.f;
  ridx[tid] = r_space[(size_t)b * A_ + tid];

  for (int e = tid; e < DP; e += 256) wb[e] = (bf16)((e < D_) ? w_rel[e] : 0.f);

  for (int e = tid; e < Q_ * DP; e += 256) {
    int q = e / DP, d = e - q * DP;
    float v = (d < D_) ? q_t[((size_t)b * Q_ + q) * D_ + d] : 0.f;
    bf16 bv = (bf16)v;
    qb[q * DP + d]  = bv;
    qTb[d * Q_ + q] = bv;
  }

  // gather relation embeddings (L2-resident table) -> bf16 LDS
  for (int r = wv; r < A_; r += 8) {
    const float* src = rel + (size_t)r_space[(size_t)b * A_ + r] * D_;
    for (int d = lane; d < DP; d += 32)
      qa[r * DP + d] = (bf16)((d < D_) ? src[d] : 0.f);
  }

  // base[d] = b_ac[d] + last_h[b] . W_ac[:D, d]   (action-independent half of the classifier)
  {
    const float* lh = last_h + (size_t)b * D_;
    for (int d = tid; d < D_; d += 256) {
      float s = b_ac[d];
      for (int e = 0; e < D_; ++e) s += lh[e] * W_ac[(size_t)e * D_ + d];
      basev[d] = s;
    }
  }
  __syncthreads();

  // ---------------- Phase 1: logits = (r_emb * w_rel) @ q_t^T  [256x32] ----------------
  {
    const float brel = b_rel[0];
    v8f acc[2][2];
#pragma unroll
    for (int t = 0; t < 2; ++t)
#pragma unroll
      for (int n = 0; n < 2; ++n)
#pragma unroll
        for (int i = 0; i < 8; ++i) acc[t][n][i] = 0.f;

    for (int kc = 0; kc < 10; ++kc) {
      const int k0 = kc * 32;
      const bf16* wp = wb + k0 + 8 * hi;
      v16bf af[2];
#pragma unroll
      for (int t = 0; t < 2; ++t) {
        v16bf f = frag_a_lds(qa, wv * 32 + t * 16, k0, DP, lane);
#pragma unroll
        for (int i = 0; i < 8; ++i) {   // scale A-fragment by w_rel (same K pattern)
          f[i]     = (bf16)((float)f[i]     * (float)wp[i]);
          f[8 + i] = (bf16)((float)f[8 + i] * (float)wp[16 + i]);
        }
        af[t] = f;
      }
      v16bf bf0 = frag_b_lds(qb, 0,  k0, DP, lane);   // B(k,n) = q_t[n][k]
      v16bf bf1 = frag_b_lds(qb, 16, k0, DP, lane);
#pragma unroll
      for (int t = 0; t < 2; ++t) {
        acc[t][0] = __builtin_amdgcn_wmma_f32_16x16x32_bf16(false, af[t], false, bf0,
                                                            (short)0, acc[t][0], false, false);
        acc[t][1] = __builtin_amdgcn_wmma_f32_16x16x32_bf16(false, af[t], false, bf1,
                                                            (short)0, acc[t][1], false, false);
      }
    }
#pragma unroll
    for (int t = 0; t < 2; ++t)
#pragma unroll
      for (int n = 0; n < 2; ++n)
#pragma unroll
        for (int v = 0; v < 8; ++v)
          logits[(wv * 32 + t * 16 + v + 8 * hi) * 32 + n * 16 + nl] = acc[t][n][v] + brel;
  }
  __syncthreads();

  // ---------------- Phase 2: softmax over Q (one row per thread) ----------------
  {
    const float* row = logits + tid * 32;
    float mx = row[0];
#pragma unroll
    for (int q = 1; q < 32; ++q) mx = fmaxf(mx, row[q]);
    float ex[32]; float s = 0.f;
#pragma unroll
    for (int q = 0; q < 32; ++q) { ex[q] = __expf(row[q] - mx); s += ex[q]; }
    float inv = 1.f / s;
#pragma unroll
    for (int q = 0; q < 32; ++q) attnb[tid * 32 + q] = (bf16)(ex[q] * inv);
  }
  __syncthreads();

  // ---------------- Phase 3: q_attn = attn @ q_t  [256x300], overwrite gather buffer ----------------
  {
#pragma unroll
    for (int t = 0; t < 2; ++t) {
      const int m0 = wv * 32 + t * 16;
      v16bf af = frag_a_lds(attnb, m0, 0, 32, lane);
      for (int nt = 0; nt < 20; ++nt) {
        v16bf bq = frag_b_lds(qTb, nt * 16, 0, 32, lane);   // B(k,n) = q_t[k][n]
        v8f c;
#pragma unroll
        for (int i = 0; i < 8; ++i) c[i] = 0.f;
        c = __builtin_amdgcn_wmma_f32_16x16x32_bf16(false, af, false, bq,
                                                    (short)0, c, false, false);
#pragma unroll
        for (int v = 0; v < 8; ++v)
          qa[(m0 + v + 8 * hi) * DP + nt * 16 + nl] = (bf16)c[v];
      }
    }
  }

  // ---------------- Phase 4: act = relu(q_attn @ W2 + base); score[a] += r_emb . act ----------------
  for (int nc = 0; nc < 5; ++nc) {
    const int nbase = nc * 64;
    v8f acc[2][4];
#pragma unroll
    for (int t = 0; t < 2; ++t)
#pragma unroll
      for (int n = 0; n < 4; ++n)
#pragma unroll
        for (int i = 0; i < 8; ++i) acc[t][n][i] = 0.f;

    for (int kc = 0; kc < 10; ++kc) {
      const int kA0 = kc * 32;
      __syncthreads();                       // panel reuse fence (also fences phase-3 writes)
      for (int e = tid; e < 64 * 32; e += 256) {     // stage W2 panel transposed
        int n = e & 63, k = e >> 6;
        int ka = kA0 + k, col = nbase + n;
        float v = (ka < D_ && col < D_) ? W_ac[(size_t)(D_ + ka) * D_ + col] : 0.f;
        panelT[n * 32 + k] = (bf16)v;
      }
      __syncthreads();
#pragma unroll
      for (int t = 0; t < 2; ++t) {
        v16bf af = frag_a_lds(qa, wv * 32 + t * 16, kA0, DP, lane);
#pragma unroll
        for (int nt = 0; nt < 4; ++nt) {
          v16bf bw = frag_b_lds(panelT, nt * 16, 0, 32, lane);
          acc[t][nt] = __builtin_amdgcn_wmma_f32_16x16x32_bf16(false, af, false, bw,
                                                               (short)0, acc[t][nt], false, false);
        }
      }
    }
    // fused epilogue: +base, relu, dot with r_emb (re-read from L2), reduce into scores
#pragma unroll
    for (int t = 0; t < 2; ++t) {
      const int m0 = wv * 32 + t * 16;
#pragma unroll
      for (int nt = 0; nt < 4; ++nt) {
        const int d = nbase + nt * 16 + nl;
#pragma unroll
        for (int v = 0; v < 8; ++v) {
          const int a = m0 + v + 8 * hi;
          float prod = 0.f;
          if (d < D_) {
            float act = fmaxf(acc[t][nt][v] + basev[d], 0.f);
            prod = act * rel[(size_t)ridx[a] * D_ + d];
          }
#pragma unroll
          for (int mk = 1; mk < 16; mk <<= 1) prod += __shfl_xor(prod, mk, 32);
          if (nl == 0) atomicAdd(&s_sc[a], prod);   // single sequential writer per address
        }
      }
    }
  }
  __syncthreads();

  // ---------------- Phase 5: softmax over A, mask, +eps, renormalize ----------------
  {
    float sc = s_sc[tid];
    float mx = block_max(sc, red, tid);
    float e  = __expf(sc - mx);
    float S  = block_sum(e, red, tid);
    float p  = e / S;
    p = p * amask[(size_t)b * A_ + tid] + EPSF;
    float S2 = block_sum(p, red, tid);
    out[(size_t)b * A_ + tid] = p / S2;
  }
}

extern "C" void kernel_launch(void* const* d_in, const int* in_sizes, int n_in,
                              void* d_out, int out_size, void* d_ws, size_t ws_size,
                              hipStream_t stream) {
  (void)in_sizes; (void)n_in; (void)d_ws; (void)ws_size; (void)out_size;
  const float* q_t   = (const float*)d_in[0];
  const float* lh    = (const float*)d_in[1];
  const float* rel   = (const float*)d_in[2];
  const float* w_rel = (const float*)d_in[3];
  const float* b_rel = (const float*)d_in[4];
  const float* W_ac  = (const float*)d_in[5];
  const float* b_ac  = (const float*)d_in[6];
  const int*   rsp   = (const int*)d_in[7];
  const float* am    = (const float*)d_in[8];
  float* out = (float*)d_out;

  hipFuncSetAttribute((const void*)srn_policy_kernel,
                      hipFuncAttributeMaxDynamicSharedMemorySize, SMEM_BYTES);
  srn_policy_kernel<<<B_, 256, SMEM_BYTES, stream>>>(q_t, lh, rel, w_rel, b_rel,
                                                     W_ac, b_ac, rsp, am, out);
}
